// MambaMixer_81578608820272
// MI455X (gfx1250) — matrix operand order repaired and checked by reading, not compiled, not run on hardware
//
#include <hip/hip_runtime.h>
#include <hip/hip_bf16.h>
#include <math.h>
#include <stdint.h>

#define HIDDEN  2048
#define INTER   4096
#define STATE   16
#define KCONV   4
#define DT_RANK 128
#define B_SZ    2
#define L_SEQ   2048
#define TOK     (B_SZ * L_SEQ)        // 4096 tokens
#define NPROJ   (2 * INTER)           // 8192
#define NSSM    (DT_RANK + 2 * STATE) // 160

typedef __bf16 bf16_t;
typedef __attribute__((ext_vector_type(16))) __bf16        bf16x16;
typedef __attribute__((ext_vector_type(8)))  __bf16        bf16x8;
typedef __attribute__((ext_vector_type(8)))  float         f32x8;
typedef __attribute__((ext_vector_type(4)))  unsigned int  u32x4;
typedef __attribute__((ext_vector_type(8)))  int           i32x8;
typedef __attribute__((ext_vector_type(4)))  int           i32x4;

// ---------------------------------------------------------------------------
// TDM: async 2-D bf16 tile load Global -> LDS via Tensor Data Mover.
// D# per CDNA5 ISA. Padding: 16 B inserted every 64 B so
// each 32-K sub-block occupies 80 B -> bank-conflict-free fragment reads.
// Tracked by TENSORcnt.
// ---------------------------------------------------------------------------
__device__ inline uint32_t lds_byte_off(const void* p) {
  // generic shared address = {aperture, lds_offset}; low 32 bits = LDS offset
  return (uint32_t)(uintptr_t)p;
}

__device__ inline void tdm_load_2d_bf16(uint32_t lds_off, const bf16_t* gptr,
                                        uint32_t tensor_w_elems,  // row length (K)
                                        uint32_t tensor_h_rows,   // total rows
                                        uint32_t tile_w_elems,    // BK
                                        uint32_t tile_h_rows) {   // BM / BN
  const uint64_t ga = (uint64_t)(uintptr_t)gptr;
  u32x4 g0;
  g0[0] = 1u;                                        // count=1, user mode
  g0[1] = lds_off;                                   // lds_addr (bytes)
  g0[2] = (uint32_t)ga;                              // global_addr[31:0]
  g0[3] = (uint32_t)((ga >> 32) & 0x01FFFFFFu)       // global_addr[56:32]
          | (2u << 30);                              // type = 2 ("image")
  const uint32_t w0 = (1u << 16)                     // data_size = 2 bytes
                    | (1u << 20)                     // pad_enable
                    | (3u << 22)                     // pad_interval = 16 DW (64B)
                    | (3u << 25);                    // pad_amount  = 4 DW (16B)
  i32x8 g1;
  g1[0] = (int)w0;
  g1[1] = (int)((tensor_w_elems & 0xFFFFu) << 16);   // abar=0 | dim0[15:0]
  g1[2] = (int)((tensor_w_elems >> 16) | ((tensor_h_rows & 0xFFFFu) << 16));
  g1[3] = (int)((tensor_h_rows >> 16) | (tile_w_elems << 16));   // | tile_dim0
  g1[4] = (int)(tile_h_rows & 0xFFFFu);              // tile_dim1 (tile_dim2=0)
  g1[5] = (int)tensor_w_elems;                       // dim0_stride[31:0] = K
  g1[6] = 0;                                         // stride hi / dim1_stride
  g1[7] = 0;
  i32x4 gz = {0, 0, 0, 0};
#if __has_include(<hip/amd_detail/amd_gfx1250_TDM.h>)
  i32x8 gz8 = {0, 0, 0, 0, 0, 0, 0, 0};
  __builtin_amdgcn_tensor_load_to_lds(g0, g1, gz, gz, gz8, 0);   // clang-23 form
#else
  __builtin_amdgcn_tensor_load_to_lds(g0, g1, gz, gz, 0);        // ROCm 7.2 form
#endif
}

// ---------------------------------------------------------------------------
// fp32 -> bf16 conversion (memory-bound, one pass per tensor)
// ---------------------------------------------------------------------------
__global__ __launch_bounds__(256)
void cvt_f32_to_bf16(const float* __restrict__ src, bf16_t* __restrict__ dst,
                     size_t n) {
  size_t i = (size_t)blockIdx.x * blockDim.x + threadIdx.x;
  if (i < n) dst[i] = (bf16_t)src[i];
}

// strided extract (dt_lr slice of ssm_p) with fp32 -> bf16
__global__ __launch_bounds__(256)
void extract_cvt_bf16(const float* __restrict__ src, bf16_t* __restrict__ dst,
                      int rows, int cols, int src_stride, int src_off) {
  size_t i = (size_t)blockIdx.x * blockDim.x + threadIdx.x;
  if (i >= (size_t)rows * cols) return;
  int r = (int)(i / cols);
  int c = (int)(i % cols);
  dst[i] = (bf16_t)src[(size_t)r * src_stride + src_off + c];
}

// ---------------------------------------------------------------------------
// bf16 WMMA GEMM:  C[M,N] = A[M,K] * B[N,K]^T   (B row-major over K)
// 256 threads = 8 waves (4M x 2N). TDM double-buffered LDS staging, BK=64,
// 8 x v_wmma_f32_16x16x32_bf16 per barrier. Wave 0 drives the TDM and waits
// on TENSORcnt; the workgroup barrier publishes the buffer to all waves.
// EPI: 0 = store f32, 1 = store bf16, 2 = +bias, softplus, store f32 (delta)
// ---------------------------------------------------------------------------
template<int BM, int BN, int EPI>
__global__ __launch_bounds__(256, 2)
void gemm_bf16_wmma(const bf16_t* __restrict__ A,
                    const bf16_t* __restrict__ Bw,
                    void* __restrict__ Cp,
                    const float* __restrict__ bias,
                    int M, int N, int K) {
  constexpr int BK    = 64;
  constexpr int LDSUB = 40;        // bf16 elems per padded 32-K sub-block (80 B)
  constexpr int LDROW = 2 * LDSUB; // 160 B per tile row (TDM pad layout)
  constexpr int WM = BM / 4;
  constexpr int WN = BN / 2;
  constexpr int SM = WM / 16;
  constexpr int SN = WN / 16;

  __shared__ bf16_t As[2][BM * LDROW];
  __shared__ bf16_t Bs[2][BN * LDROW];

  const int tid   = threadIdx.x;
  const int wave  = tid >> 5;
  const int lane  = tid & 31;
  const int wm    = wave & 3;
  const int wn    = wave >> 2;
  const int tileM = blockIdx.y * BM;
  const int tileN = blockIdx.x * BN;
  const bool loader = (wave == 0);

  f32x8 acc[SM][SN];
#pragma unroll
  for (int i = 0; i < SM; ++i)
#pragma unroll
    for (int j = 0; j < SN; ++j)
#pragma unroll
      for (int r = 0; r < 8; ++r) acc[i][j][r] = 0.0f;

  const int am    = lane & 15;   // row (A) / col (B) within 16
  const int half  = lane >> 4;   // lane half selects K sub-range per ISA layout
  const int khalf = half * 8;    // A frag: K = khalf..khalf+7, 16+khalf..
  const int kb    = half * 16;   // B frag: same K order as A (khalf..khalf+7, 16+khalf..)

  // prologue: kick off tile 0 into buffer 0
  if (loader) {
    tdm_load_2d_bf16(lds_byte_off(&As[0][0]), &A[(size_t)tileM * K],
                     (uint32_t)K, (uint32_t)M, BK, BM);
    tdm_load_2d_bf16(lds_byte_off(&Bs[0][0]), &Bw[(size_t)tileN * K],
                     (uint32_t)K, (uint32_t)N, BK, BN);
  }

  int cur = 0;
  for (int k0 = 0; k0 < K; k0 += BK) {
    if (loader) __builtin_amdgcn_s_wait_tensorcnt(0);
    __syncthreads();                       // buffer 'cur' now visible to all

    if (loader && (k0 + BK < K)) {         // overlap next TDM with compute
      const int nxt = cur ^ 1;
      tdm_load_2d_bf16(lds_byte_off(&As[nxt][0]),
                       &A[(size_t)tileM * K + k0 + BK],
                       (uint32_t)K, (uint32_t)M, BK, BM);
      tdm_load_2d_bf16(lds_byte_off(&Bs[nxt][0]),
                       &Bw[(size_t)tileN * K + k0 + BK],
                       (uint32_t)K, (uint32_t)N, BK, BN);
    }

#pragma unroll
    for (int ks = 0; ks < 2; ++ks) {       // two 32-K sub-blocks per buffer
      bf16x16 afrag[SM];
#pragma unroll
      for (int i = 0; i < SM; ++i) {
        const bf16_t* ap =
            &As[cur][(wm * WM + i * 16 + am) * LDROW + ks * LDSUB];
        bf16x8 lo = *(const bf16x8*)(ap + khalf);
        bf16x8 hi = *(const bf16x8*)(ap + 16 + khalf);
        afrag[i] = __builtin_shufflevector(lo, hi, 0,1,2,3,4,5,6,7,
                                                   8,9,10,11,12,13,14,15);
      }
      bf16x16 bfrag[SN];
#pragma unroll
      for (int j = 0; j < SN; ++j) {
        const bf16_t* bp =
            &Bs[cur][(wn * WN + j * 16 + am) * LDROW + ks * LDSUB];
        bf16x8 lo = *(const bf16x8*)(bp + khalf);
        bf16x8 hi = *(const bf16x8*)(bp + 16 + khalf);
        bfrag[j] = __builtin_shufflevector(lo, hi, 0,1,2,3,4,5,6,7,
                                                   8,9,10,11,12,13,14,15);
      }
#pragma unroll
      for (int i = 0; i < SM; ++i)
#pragma unroll
        for (int j = 0; j < SN; ++j)
          acc[i][j] = __builtin_amdgcn_wmma_f32_16x16x32_bf16(
              false, afrag[i], false, bfrag[j], (short)0, acc[i][j],
              false, false);
    }
    cur ^= 1;
  }

  // epilogue; C layout: VGPR r -> M = r + 8*(lane>=16), N = lane&15
  const int rhalf = half * 8;
#pragma unroll
  for (int i = 0; i < SM; ++i) {
#pragma unroll
    for (int j = 0; j < SN; ++j) {
      const int col = tileN + wn * WN + j * 16 + am;
#pragma unroll
      for (int r = 0; r < 8; ++r) {
        const int row = tileM + wm * WM + i * 16 + rhalf + r;
        float v = acc[i][j][r];
        if constexpr (EPI == 2) {
          v += bias[col];
          v = (v > 20.0f) ? v : log1pf(__expf(v));   // softplus
        }
        if constexpr (EPI == 1) {
          ((bf16_t*)Cp)[(size_t)row * N + col] = (bf16_t)v;
        } else {
          ((float*)Cp)[(size_t)row * N + col] = v;
        }
      }
    }
  }
}

// ---------------------------------------------------------------------------
// causal depthwise conv (K=4) + bias + SiLU; x = proj[:, :INTER]
// ---------------------------------------------------------------------------
__global__ __launch_bounds__(256)
void conv_silu(const bf16_t* __restrict__ proj, const float* __restrict__ cw,
               const float* __restrict__ cb, bf16_t* __restrict__ u) {
  const int d = blockIdx.x * 256 + threadIdx.x;  // channel
  const int l = blockIdx.y;                      // seq pos
  const int b = blockIdx.z;                      // batch
  const size_t rs = NPROJ;
  const bf16_t* xcol = proj + (size_t)b * L_SEQ * rs + d;
  float a = cb[d];
#pragma unroll
  for (int k = 0; k < KCONV; ++k) {
    const int ls = l - (KCONV - 1) + k;
    if (ls >= 0) a += (float)xcol[(size_t)ls * rs] * cw[d * KCONV + k];
  }
  const float s = a / (1.0f + __expf(-a));       // SiLU
  u[((size_t)(b * L_SEQ + l)) * INTER + d] = (bf16_t)s;
}

// ---------------------------------------------------------------------------
// selective scan: one wave handles 2 channels; lanes 0-15 / 16-31 hold the
// 16 SSM states of each channel. __shfl_xor(1,2,4,8) reduces h*C over states.
// Fused epilogue: y = (scan + u*D) * silu(gate) -> bf16 for out_proj GEMM.
// ---------------------------------------------------------------------------
__global__ __launch_bounds__(256)
void selective_scan(const float* __restrict__ delta,
                    const bf16_t* __restrict__ u,
                    const float* __restrict__ ssmp,
                    const bf16_t* __restrict__ proj,   // gate = cols INTER..
                    const float* __restrict__ A_log,
                    const float* __restrict__ Dp,
                    bf16_t* __restrict__ y) {
  const int b    = blockIdx.y;
  const int wave = threadIdx.x >> 5;
  const int lane = threadIdx.x & 31;
  const int n    = lane & 15;
  const int d    = blockIdx.x * 16 + wave * 2 + (lane >> 4);

  const float Adn = -__expf(A_log[d * STATE + n]);
  const float Dd  = Dp[d];
  float h = 0.0f;

  for (int l = 0; l < L_SEQ; ++l) {
    const size_t tok = (size_t)b * L_SEQ + l;
    const float dlt = delta[tok * INTER + d];
    const float ut  = (float)u[tok * INTER + d];
    const float Bt  = ssmp[tok * NSSM + DT_RANK + n];
    const float Ct  = ssmp[tok * NSSM + DT_RANK + STATE + n];

    h = h * __expf(dlt * Adn) + dlt * ut * Bt;
    float yv = h * Ct;
    yv += __shfl_xor(yv, 1);
    yv += __shfl_xor(yv, 2);
    yv += __shfl_xor(yv, 4);
    yv += __shfl_xor(yv, 8);

    if (n == 0) {
      const float g  = (float)proj[tok * NPROJ + INTER + d];
      const float sg = g / (1.0f + __expf(-g));
      y[tok * INTER + d] = (bf16_t)((yv + ut * Dd) * sg);
    }
  }
}

// ---------------------------------------------------------------------------
extern "C" void kernel_launch(void* const* d_in, const int* in_sizes, int n_in,
                              void* d_out, int out_size, void* d_ws,
                              size_t ws_size, hipStream_t stream) {
  (void)in_sizes; (void)n_in; (void)out_size; (void)ws_size;

  const float* hs     = (const float*)d_in[0];  // (B,L,H)
  const float* w_in   = (const float*)d_in[1];  // (2E,H)
  const float* conv_w = (const float*)d_in[2];  // (E,1,4)
  const float* conv_b = (const float*)d_in[3];  // (E)
  const float* w_xp   = (const float*)d_in[4];  // (160,E)
  const float* w_dt   = (const float*)d_in[5];  // (E,128)
  const float* dt_b   = (const float*)d_in[6];  // (E)
  const float* A_log  = (const float*)d_in[7];  // (E,16)
  const float* Dp     = (const float*)d_in[8];  // (E)
  const float* w_out  = (const float*)d_in[9];  // (H,E)
  float* out = (float*)d_out;                   // (B,L,H) f32

  char* base = (char*)d_ws;
  size_t off = 0;
  auto take = [&](size_t bytes) -> void* {
    void* p = base + off;
    off += (bytes + 255) & ~(size_t)255;
    return p;
  };
  bf16_t* hs_bf    = (bf16_t*)take((size_t)TOK * HIDDEN * 2);
  bf16_t* w_in_bf  = (bf16_t*)take((size_t)NPROJ * HIDDEN * 2);
  bf16_t* w_xp_bf  = (bf16_t*)take((size_t)NSSM * INTER * 2);
  bf16_t* w_dt_bf  = (bf16_t*)take((size_t)INTER * DT_RANK * 2);
  bf16_t* w_out_bf = (bf16_t*)take((size_t)HIDDEN * INTER * 2);
  bf16_t* proj_bf  = (bf16_t*)take((size_t)TOK * NPROJ * 2);
  bf16_t* u_bf     = (bf16_t*)take((size_t)TOK * INTER * 2);
  float*  ssmp     = (float*) take((size_t)TOK * NSSM * 4);
  bf16_t* dt_bf    = (bf16_t*)take((size_t)TOK * DT_RANK * 2);
  float*  delta    = (float*) take((size_t)TOK * INTER * 4);
  bf16_t* y_bf     = (bf16_t*)take((size_t)TOK * INTER * 2);

  auto cvt = [&](const float* s, bf16_t* d, size_t n) {
    cvt_f32_to_bf16<<<(unsigned)((n + 255) / 256), 256, 0, stream>>>(s, d, n);
  };
  cvt(hs,    hs_bf,    (size_t)TOK * HIDDEN);
  cvt(w_in,  w_in_bf,  (size_t)NPROJ * HIDDEN);
  cvt(w_xp,  w_xp_bf,  (size_t)NSSM * INTER);
  cvt(w_dt,  w_dt_bf,  (size_t)INTER * DT_RANK);
  cvt(w_out, w_out_bf, (size_t)HIDDEN * INTER);

  // 1) in_proj: (4096x2048) x (8192x2048)^T -> proj bf16 (4096x8192)
  gemm_bf16_wmma<128, 64, 1>
      <<<dim3(NPROJ / 64, TOK / 128), 256, 0, stream>>>(
          hs_bf, w_in_bf, proj_bf, nullptr, TOK, NPROJ, HIDDEN);

  // 2) depthwise causal conv + SiLU -> u bf16
  conv_silu<<<dim3(INTER / 256, L_SEQ, B_SZ), 256, 0, stream>>>(
      proj_bf, conv_w, conv_b, u_bf);

  // 3) x_proj: (4096x4096) x (160x4096)^T -> ssm_p f32 (4096x160)
  gemm_bf16_wmma<128, 32, 0>
      <<<dim3(NSSM / 32, TOK / 128), 256, 0, stream>>>(
          u_bf, w_xp_bf, ssmp, nullptr, TOK, NSSM, INTER);

  // 4) dt_lr slice -> bf16
  {
    size_t n = (size_t)TOK * DT_RANK;
    extract_cvt_bf16<<<(unsigned)((n + 255) / 256), 256, 0, stream>>>(
        ssmp, dt_bf, TOK, DT_RANK, NSSM, 0);
  }

  // 5) dt_proj + bias + softplus -> delta f32 (4096x4096)
  gemm_bf16_wmma<128, 64, 2>
      <<<dim3(INTER / 64, TOK / 128), 256, 0, stream>>>(
          dt_bf, w_dt_bf, delta, dt_b, TOK, INTER, DT_RANK);

  // 6) selective scan + D skip + silu(gate) -> y bf16
  selective_scan<<<dim3(INTER / 16, B_SZ), 256, 0, stream>>>(
      delta, u_bf, ssmp, proj_bf, A_log, Dp, y_bf);

  // 7) out_proj: (4096x4096) x (2048x4096)^T -> out f32 (4096x2048)
  gemm_bf16_wmma<128, 64, 0>
      <<<dim3(HIDDEN / 64, TOK / 128), 256, 0, stream>>>(
          y_bf, w_out_bf, out, nullptr, TOK, HIDDEN, INTER);
}
